// MGraphDTA_90640989815116
// MI455X (gfx1250) — compile-verified, matrix operand order checked by reading, not run on hardware
//
#include <hip/hip_runtime.h>
#include <hip/hip_bf16.h>

// ---------------------------------------------------------------------------
// MGraphDTA forward for MI455X (gfx1250, wave32, WMMA).
// GEMMs: bf16 WMMA (v_wmma_f32_16x16x32_bf16), f32 accumulate, 32x32 tile per
// wave (4 WMMA/K-step), B128 staging through LDS, async-to-LDS on fast path.
// GAT branch: float atomics + order-preserving uint atomicMax.
// ---------------------------------------------------------------------------

#define USE_ASYNC_LDS 1

#define B_ 8
#define S_ 1200
#define PIN_ 1024
#define EMB_ 128
#define FNUM_ 96
#define N_ 320
#define E_ 4096
#define EDGES_ (E_ + N_)
#define MAXN_ 40

typedef __attribute__((ext_vector_type(16))) __bf16 v16bf;
typedef __attribute__((ext_vector_type(8)))  float  v8f;

static __device__ __forceinline__ unsigned short f2bf(float f) {
  unsigned u = __float_as_uint(f);
  u += 0x7FFFu + ((u >> 16) & 1u);          // round-to-nearest-even
  return (unsigned short)(u >> 16);
}

union BFV { v16bf v; unsigned short u[16]; };

// ---------------------------------------------------------------------------
// Batched GEMM: C[b] = act(alpha * A[b] @ B[b](^T) + bias)
// One wave32 per 32x32 output tile, K stepped by 32.
// ---------------------------------------------------------------------------
__global__ __launch_bounds__(32)
void gemm_wmma(const float* __restrict__ A, long long strideA, int lda,
               const float* __restrict__ Bm, long long strideB, int ldb, int transB,
               float* __restrict__ C, long long strideC, int ldc,
               const float* __restrict__ bias,
               int M, int N, int K, float alpha, int act)
{
  __shared__ float sA[32][36];              // padded: rows stay 16B aligned
  __shared__ float sB[32][36];
  const int tile_n = blockIdx.x * 32;
  const int tile_m = blockIdx.y * 32;
  A  += (long long)blockIdx.z * strideA;
  Bm += (long long)blockIdx.z * strideB;
  C  += (long long)blockIdx.z * strideC;
  const int lane = threadIdx.x;
  const int mrow = lane & 15;
  const int hi   = lane >> 4;
  const bool vecA = ((lda & 3) == 0);
  const bool vecB = ((ldb & 3) == 0);
  v8f acc00 = {}, acc01 = {}, acc10 = {}, acc11 = {};

  for (int kk = 0; kk < K; kk += 32) {
    const bool fullK = (kk + 32 <= K);
    const bool fastA = fullK && vecA && (tile_m + 32 <= M);
    const bool fastB = fullK && vecB && (tile_n + 32 <= N);

    // ---------------- stage A tile (32 x 32) ----------------
    if (fastA) {
      #pragma unroll
      for (int i = 0; i < 8; ++i) {
        int f = i * 32 + lane;
        int r = f >> 3, c4 = (f & 7) * 4;
        const float* gp = A + (long long)(tile_m + r) * lda + kk + c4;
#if USE_ASYNC_LDS
        unsigned lo = (unsigned)(unsigned long long)&sA[r][c4];
        asm volatile("global_load_async_to_lds_b128 %0, %1, off"
                     :: "v"(lo), "v"(gp) : "memory");
#else
        *(float4*)&sA[r][c4] = *(const float4*)gp;
#endif
      }
    } else {
      for (int i = 0; i < 32; ++i) {
        int f = i * 32 + lane;
        int r = f >> 5, c = f & 31;
        int gm = tile_m + r, gk = kk + c;
        sA[r][c] = (gm < M && gk < K) ? A[(long long)gm * lda + gk] : 0.f;
      }
    }

    // ---------------- stage B tile (32 x 32) ----------------
    if (fastB && !transB) {
      #pragma unroll
      for (int i = 0; i < 8; ++i) {
        int f = i * 32 + lane;
        int r = f >> 3, c4 = (f & 7) * 4;     // r = k, c4 = n offset
        *(float4*)&sB[r][c4] =
            *(const float4*)(Bm + (long long)(kk + r) * ldb + tile_n + c4);
      }
    } else if (fastB && transB) {
      #pragma unroll
      for (int i = 0; i < 8; ++i) {
        int f = i * 32 + lane;
        int r = f >> 3, c4 = (f & 7) * 4;     // r = n row, c4 = k offset
        float4 vv = *(const float4*)(Bm + (long long)(tile_n + r) * ldb + kk + c4);
        sB[c4 + 0][r] = vv.x;
        sB[c4 + 1][r] = vv.y;
        sB[c4 + 2][r] = vv.z;
        sB[c4 + 3][r] = vv.w;
      }
    } else {
      for (int i = 0; i < 32; ++i) {
        int f = i * 32 + lane;
        int r = f >> 5, c = f & 31;           // r = k, c = n
        int gk = kk + r, gn = tile_n + c;
        float v = 0.f;
        if (gk < K && gn < N)
          v = transB ? Bm[(long long)gn * ldb + gk] : Bm[(long long)gk * ldb + gn];
        sB[r][c] = v;
      }
    }

    // prefetch next K slice (speculative global_prefetch)
    if (kk + 32 < K) {
      __builtin_prefetch(A + (long long)(tile_m + (lane & 31)) * lda + kk + 32, 0, 0);
      if (!transB)
        __builtin_prefetch(Bm + (long long)(kk + 32 + (lane & 31)) * ldb + tile_n, 0, 0);
    }

#if USE_ASYNC_LDS
    if (fastA) asm volatile("s_wait_asynccnt 0" ::: "memory");
#endif
    __syncthreads();

    // fragments per CDNA5 16-bit 16x32 layout (ISA 7.12.2)
    BFV a0, a1, b0, b1;
    #pragma unroll
    for (int j = 0; j < 8; ++j) {
      int kb = (j < 4) ? (8 * hi + 2 * j) : (16 + 8 * hi + 2 * (j - 4));
      a0.u[2 * j]     = f2bf(sA[mrow][kb]);
      a0.u[2 * j + 1] = f2bf(sA[mrow][kb + 1]);
      a1.u[2 * j]     = f2bf(sA[16 + mrow][kb]);
      a1.u[2 * j + 1] = f2bf(sA[16 + mrow][kb + 1]);
      b0.u[2 * j]     = f2bf(sB[kb][mrow]);
      b0.u[2 * j + 1] = f2bf(sB[kb + 1][mrow]);
      b1.u[2 * j]     = f2bf(sB[kb][16 + mrow]);
      b1.u[2 * j + 1] = f2bf(sB[kb + 1][16 + mrow]);
    }
    acc00 = __builtin_amdgcn_wmma_f32_16x16x32_bf16(false, a0.v, false, b0.v, (short)0, acc00, false, false);
    acc01 = __builtin_amdgcn_wmma_f32_16x16x32_bf16(false, a0.v, false, b1.v, (short)0, acc01, false, false);
    acc10 = __builtin_amdgcn_wmma_f32_16x16x32_bf16(false, a1.v, false, b0.v, (short)0, acc10, false, false);
    acc11 = __builtin_amdgcn_wmma_f32_16x16x32_bf16(false, a1.v, false, b1.v, (short)0, acc11, false, false);
    __syncthreads();
  }

  // C/D layout: n = lane&15, m = r + 8*(lane>>4)
  const int n0 = tile_n + mrow;
  const int n1 = n0 + 16;
  #pragma unroll
  for (int r = 0; r < 8; ++r) {
    int m0 = tile_m + r + 8 * hi;
    int m1 = m0 + 16;
    float v;
    if (m0 < M && n0 < N) {
      v = acc00[r] * alpha; if (bias) v += bias[n0]; if (act == 1) v = fmaxf(v, 0.f);
      C[(long long)m0 * ldc + n0] = v;
    }
    if (m0 < M && n1 < N) {
      v = acc01[r] * alpha; if (bias) v += bias[n1]; if (act == 1) v = fmaxf(v, 0.f);
      C[(long long)m0 * ldc + n1] = v;
    }
    if (m1 < M && n0 < N) {
      v = acc10[r] * alpha; if (bias) v += bias[n0]; if (act == 1) v = fmaxf(v, 0.f);
      C[(long long)m1 * ldc + n0] = v;
    }
    if (m1 < M && n1 < N) {
      v = acc11[r] * alpha; if (bias) v += bias[n1]; if (act == 1) v = fmaxf(v, 0.f);
      C[(long long)m1 * ldc + n1] = v;
    }
  }
}

// conv1d k=3 pad=1 + bias + relu, layouts [B,S,Cin] -> [B,S,Cout]
__global__ void conv1d_relu(const float* __restrict__ x, const float* __restrict__ W,
                            const float* __restrict__ bias, float* __restrict__ y,
                            int S, int Cin, int Cout)
{
  extern __shared__ float sx[];           // 3*Cin
  int s = blockIdx.x, b = blockIdx.y, o = threadIdx.x;
  for (int i = o; i < 3 * Cin; i += blockDim.x) {
    int t = i / Cin, ci = i - t * Cin;
    int ss = s + t - 1;
    sx[i] = (ss >= 0 && ss < S) ? x[((long long)b * S + ss) * Cin + ci] : 0.f;
  }
  __syncthreads();
  float acc = bias[o];
  for (int t = 0; t < 3; ++t)
    for (int ci = 0; ci < Cin; ++ci)
      acc += W[(o * Cin + ci) * 3 + t] * sx[t * Cin + ci];
  y[((long long)b * S + s) * Cout + o] = fmaxf(acc, 0.f);
}

// leaky_relu(0.01) then LayerNorm over channel dim (96) with affine
__global__ void leaky_ln(const float* __restrict__ x, const float* __restrict__ g,
                         const float* __restrict__ bt, float* __restrict__ y,
                         int S, int C)
{
  __shared__ float sv[FNUM_];
  __shared__ float smu, srs;
  int s = blockIdx.x, b = blockIdx.y, c = threadIdx.x;
  float v = x[((long long)b * S + s) * C + c];
  v = v > 0.f ? v : 0.01f * v;
  sv[c] = v;
  __syncthreads();
  if (c == 0) {
    float m = 0.f;
    for (int i = 0; i < C; ++i) m += sv[i];
    m /= C;
    float va = 0.f;
    for (int i = 0; i < C; ++i) { float d = sv[i] - m; va += d * d; }
    va /= C;
    smu = m; srs = rsqrtf(va + 1e-5f);
  }
  __syncthreads();
  y[((long long)b * S + s) * C + c] = (v - smu) * srs * g[c] + bt[c];
}

// row softmax with additive key mask; rows s[b][q][0..L)
__global__ void softmax_rows(float* __restrict__ s, const float* __restrict__ mask,
                             int Q, int L)
{
  __shared__ float red[256];
  int q = blockIdx.x, b = blockIdx.y, t = threadIdx.x;
  float* row = s + ((long long)b * Q + q) * L;
  const float* mrow = mask ? mask + (long long)b * L : nullptr;
  float mx = -3.4e38f;
  for (int i = t; i < L; i += 256) {
    float v = row[i] + (mrow ? mrow[i] : 0.f);
    row[i] = v;
    mx = fmaxf(mx, v);
  }
  red[t] = mx; __syncthreads();
  for (int o = 128; o > 0; o >>= 1) { if (t < o) red[t] = fmaxf(red[t], red[t + o]); __syncthreads(); }
  mx = red[0]; __syncthreads();
  float sum = 0.f;
  for (int i = t; i < L; i += 256) { float e = expf(row[i] - mx); row[i] = e; sum += e; }
  red[t] = sum; __syncthreads();
  for (int o = 128; o > 0; o >>= 1) { if (t < o) red[t] += red[t + o]; __syncthreads(); }
  float inv = 1.f / red[0];
  for (int i = t; i < L; i += 256) row[i] *= inv;
}

// seq[b,c,s] = (xs*w0 + xa*w1)[b,s,c] with w = mw/sum(mw)
__global__ void mix_transpose(const float* __restrict__ xs, const float* __restrict__ xa,
                              const float* __restrict__ mw, float* __restrict__ out,
                              int S, int C)
{
  long long i = (long long)blockIdx.x * blockDim.x + threadIdx.x;
  long long total = (long long)B_ * C * S;
  if (i >= total) return;
  int s = (int)(i % S); long long r = i / S;
  int c = (int)(r % C); int b = (int)(r / C);
  float w0 = mw[0], w1 = mw[1], inv = 1.f / (w0 + w1);
  long long src = ((long long)b * S + s) * C + c;
  out[((long long)b * C + c) * S + s] = xs[src] * w0 * inv + xa[src] * w1 * inv;
}

__global__ void wsum2(const float* __restrict__ a, const float* __restrict__ b,
                      const float* __restrict__ w, float* __restrict__ out, long long n)
{
  long long i = (long long)blockIdx.x * blockDim.x + threadIdx.x;
  if (i >= n) return;
  float w0 = w[0], w1 = w[1], inv = 1.f / (w0 + w1);
  out[i] = a[i] * w0 * inv + b[i] * w1 * inv;
}

// [B,C,S] -> [B,S,C]
__global__ void transpose_cs(const float* __restrict__ in, float* __restrict__ out,
                             int S, int C)
{
  long long i = (long long)blockIdx.x * blockDim.x + threadIdx.x;
  long long total = (long long)B_ * S * C;
  if (i >= total) return;
  int c = (int)(i % C); long long r = i / C;
  int s = (int)(r % S); int b = (int)(r / S);
  out[((long long)b * S + s) * C + c] = in[((long long)b * C + c) * S + s];
}

// ---------------- GAT support ----------------
static __device__ __forceinline__ unsigned fkey(float f) {
  unsigned u = __float_as_uint(f);
  return (u & 0x80000000u) ? ~u : (u | 0x80000000u);
}
static __device__ __forceinline__ float funkey(unsigned k) {
  unsigned u = (k & 0x80000000u) ? (k & 0x7FFFFFFFu) : ~k;
  return __uint_as_float(u);
}

__global__ void gat_alpha(const float* __restrict__ H, int ldh,
                          const float* __restrict__ as_, const float* __restrict__ ad_,
                          float* __restrict__ asrc, float* __restrict__ adst,
                          int Nn, int F)
{
  int i = blockIdx.x * blockDim.x + threadIdx.x;
  if (i >= Nn) return;
  float s = 0.f, d = 0.f;
  for (int f = 0; f < F; ++f) { float h = H[(long long)i * ldh + f]; s += h * as_[f]; d += h * ad_[f]; }
  asrc[i] = s; adst[i] = d;
}

__global__ void gat_init(unsigned* __restrict__ mkey, float* __restrict__ den,
                         float* __restrict__ outacc, int ldo, int Nn, int F)
{
  int i = blockIdx.x * blockDim.x + threadIdx.x;
  if (i < Nn) { mkey[i] = 0u; den[i] = 0.f; }
  if (i < Nn * F) outacc[(long long)(i / F) * ldo + (i % F)] = 0.f;
}

__global__ void edge_max(const int* __restrict__ ei, const float* __restrict__ asrc,
                         const float* __restrict__ adst, unsigned* __restrict__ mkey)
{
  int e = blockIdx.x * blockDim.x + threadIdx.x;
  if (e >= EDGES_) return;
  int s = (e < E_) ? ei[e] : (e - E_);
  int d = (e < E_) ? ei[E_ + e] : (e - E_);
  float a = asrc[s] + adst[d];
  a = a > 0.f ? a : 0.2f * a;
  atomicMax(&mkey[d], fkey(a));
}

__global__ void edge_expsum(const int* __restrict__ ei, const float* __restrict__ asrc,
                            const float* __restrict__ adst, const unsigned* __restrict__ mkey,
                            float* __restrict__ ex, float* __restrict__ den)
{
  int e = blockIdx.x * blockDim.x + threadIdx.x;
  if (e >= EDGES_) return;
  int s = (e < E_) ? ei[e] : (e - E_);
  int d = (e < E_) ? ei[E_ + e] : (e - E_);
  float a = asrc[s] + adst[d];
  a = a > 0.f ? a : 0.2f * a;
  float v = expf(a - funkey(mkey[d]));
  ex[e] = v;
  atomicAdd(&den[d], v);
}

__global__ void edge_accum(const int* __restrict__ ei, const float* __restrict__ ex,
                           const float* __restrict__ den, const float* __restrict__ H,
                           int ldh, float* __restrict__ outacc, int ldo, int F)
{
  int idx = blockIdx.x * blockDim.x + threadIdx.x;
  if (idx >= EDGES_ * F) return;
  int e = idx / F, f = idx - e * F;
  int s = (e < E_) ? ei[e] : (e - E_);
  int d = (e < E_) ? ei[E_ + e] : (e - E_);
  atomicAdd(&outacc[(long long)d * ldo + f], (ex[e] / den[d]) * H[(long long)s * ldh + f]);
}

// per-feature batch-norm over nodes (training stats) + bias + affine + relu
__global__ void gat_bn_relu(const float* __restrict__ outacc, int ldo,
                            const float* __restrict__ bias, const float* __restrict__ g,
                            const float* __restrict__ be, float* __restrict__ dst,
                            int ldd, int Nn)
{
  __shared__ float red[64];
  int f = blockIdx.x, t = threadIdx.x;
  float s = 0.f;
  for (int i = t; i < Nn; i += 64) s += outacc[(long long)i * ldo + f] + bias[f];
  red[t] = s; __syncthreads();
  for (int o = 32; o > 0; o >>= 1) { if (t < o) red[t] += red[t + o]; __syncthreads(); }
  float mu = red[0] / Nn; __syncthreads();
  float v = 0.f;
  for (int i = t; i < Nn; i += 64) { float d = outacc[(long long)i * ldo + f] + bias[f] - mu; v += d * d; }
  red[t] = v; __syncthreads();
  for (int o = 32; o > 0; o >>= 1) { if (t < o) red[t] += red[t + o]; __syncthreads(); }
  float inv = rsqrtf(red[0] / Nn + 1e-5f);
  for (int i = t; i < Nn; i += 64) {
    float val = (outacc[(long long)i * ldo + f] + bias[f] - mu) * inv * g[f] + be[f];
    dst[(long long)i * ldd + f] = fmaxf(val, 0.f);
  }
}

// dst[b*ldd + f] = mean over 40 nodes of src[(b*40+i)*96 + f]
__global__ void mean40(const float* __restrict__ src, float* __restrict__ dst, int ldd)
{
  int b = blockIdx.x, f = threadIdx.x;
  float s = 0.f;
  for (int i = 0; i < MAXN_; ++i) s += src[((long long)b * MAXN_ + i) * FNUM_ + f];
  dst[(long long)b * ldd + f] = s * (1.f / MAXN_);
}

// ---------------------------------------------------------------------------
extern "C" void kernel_launch(void* const* d_in, const int* in_sizes, int n_in,
                              void* d_out, int out_size, void* d_ws, size_t ws_size,
                              hipStream_t stream)
{
  (void)in_sizes; (void)n_in; (void)out_size; (void)ws_size;
  const float* x_in  = (const float*)d_in[0];   // [320,22]
  const int*   ei    = (const int*)d_in[1];     // [2,4096]
  const float* prot5 = (const float*)d_in[3];   // [8,1200,1024]
  const float* mask  = (const float*)d_in[4];   // [8,1200]
  auto P = [&](int i) { return (const float*)d_in[i]; };

  // ---- flattened params (JAX pytree: dict keys sorted, lists in order) ----
  // 5..12  cls[0..3]{W,b}
  // 13..20 cross {Wk,Wo,Wq,Wv,bk,bo,bq,bv}
  // 21..308 lig.blocks[3][8]{c1,c2} x gat{W,ad,as,b,be,g}
  // 309 lig.cls_W  310 lig.cls_b  311..316 lig.conv0  317..334 lig.trans[3]
  // 335.. prot.blocks: b0{attn 335..342, conv 343/344, lin 345..350, mix 351}
  //                    b1{attn 352..359, convs 360..363, lin 364..369, ln 370/371, mix 372}
  // 373 out_W 374 out_b 375 resh_W 376 resh_b 377 weights
  struct PB { int attn, c0W, c0b, c1W, c1b, lnb, lng, lin, mix; };
  const PB pb[2] = { {335, 343, 344, -1, -1, -1, -1, 345, 351},
                     {352, 360, 361, 362, 363, 370, 371, 364, 372} };

  // ---- workspace layout ----
  float* W0 = (float*)d_ws;
  size_t off = 0;
  auto alloc = [&](size_t n) { float* p = W0 + off; off += n; return p; };
  float* h      = alloc(1228800);     // [8,1200,128]
  float* buf1   = alloc(921600);
  float* buf2   = alloc(921600);
  float* qh     = alloc(921600);
  float* kh     = alloc(921600);
  float* vh     = alloc(921600);
  float* ctx    = alloc(921600);
  float* xa     = alloc(921600);
  float* scores = alloc(11520000);    // [8,1200,1200] (reused for cross [8,40,1200])
  float* seqb0  = alloc(921600);
  float* seqb1  = alloc(921600);
  float* seqfeat= alloc(921600);
  float* seqT   = alloc(921600);
  float* f1     = alloc(8192);
  float* f2     = alloc(2048);
  float* feats  = alloc(1536);        // [8,192]
  float* G1     = alloc(145920);      // [320,456]
  float* G2     = alloc(145920);
  float* gH     = alloc(81920);       // [320,256]
  float* gout   = alloc(81920);
  float* gT     = alloc(20480);       // [320,64]
  float* asrc   = alloc(320);
  float* adst   = alloc(320);
  unsigned* mkey = (unsigned*)alloc(320);
  float* den    = alloc(320);
  float* exb    = alloc(4416);
  float* nodef  = alloc(30720);       // [320,96]
  float* qh2    = alloc(30720);
  float* ctx2   = alloc(30720);
  float* ca     = alloc(30720);
  float* z      = alloc(2304);        // [8,288]
  float* t1     = alloc(8192);
  float* t2     = alloc(8192);
  float* t3     = alloc(2048);
  float* seqbs[2] = { seqb0, seqb1 };

  const float rs96 = 0.1020620726f;   // 1/sqrt(96)

  auto gemm = [&](const float* A, long long sA, int lda,
                  const float* Bm, long long sB, int ldb, int tB,
                  float* C, long long sC, int ldc, const float* bias,
                  int M, int N, int K, float alpha, int act, int batch) {
    dim3 g((N + 31) / 32, (M + 31) / 32, batch);
    gemm_wmma<<<g, dim3(32), 0, stream>>>(A, sA, lda, Bm, sB, ldb, tB,
                                          C, sC, ldc, bias, M, N, K, alpha, act);
  };

  auto run_gat = [&](const float* Ain, int lda, int inF, int outF,
                     int pbase, float* dst, int ldd) {
    // gat params (sorted): W, ad, as, b, be, g
    const float* Wm = P(pbase + 0);
    const float* ad = P(pbase + 1);
    const float* as_ = P(pbase + 2);
    const float* bb = P(pbase + 3);
    const float* be = P(pbase + 4);
    const float* gg = P(pbase + 5);
    gemm(Ain, 0, lda, Wm, 0, outF, 0, gH, 0, 256, nullptr, N_, outF, inF, 1.f, 0, 1);
    gat_alpha<<<(N_ + 63) / 64, 64, 0, stream>>>(gH, 256, as_, ad, asrc, adst, N_, outF);
    gat_init<<<(N_ * outF + 255) / 256, 256, 0, stream>>>(mkey, den, gout, 256, N_, outF);
    edge_max<<<(EDGES_ + 255) / 256, 256, 0, stream>>>(ei, asrc, adst, mkey);
    edge_expsum<<<(EDGES_ + 255) / 256, 256, 0, stream>>>(ei, asrc, adst, mkey, exb, den);
    edge_accum<<<(EDGES_ * outF + 255) / 256, 256, 0, stream>>>(ei, exb, den, gH, 256, gout, 256, outF);
    gat_bn_relu<<<outF, 64, 0, stream>>>(gout, 256, bb, gg, be, dst, ldd, N_);
  };

  // ================= protein branch =================
  // h = relu(prot5 @ resh_W + resh_b)     [9600,1024]x[1024,128]
  gemm(prot5, 0, PIN_, P(375), 0, EMB_, 0, h, 0, EMB_, P(376),
       B_ * S_, EMB_, PIN_, 1.f, 1, 1);

  for (int bi = 0; bi < 2; ++bi) {
    // conv stack (layouts [B,S,C])
    conv1d_relu<<<dim3(S_, B_), FNUM_, 3 * EMB_ * sizeof(float), stream>>>(
        h, P(pb[bi].c0W), P(pb[bi].c0b), buf1, S_, EMB_, FNUM_);
    if (bi == 1) {
      leaky_ln<<<dim3(S_, B_), FNUM_, 0, stream>>>(buf1, P(pb[bi].lng), P(pb[bi].lnb),
                                                   buf2, S_, FNUM_);
      conv1d_relu<<<dim3(S_, B_), FNUM_, 3 * FNUM_ * sizeof(float), stream>>>(
          buf2, P(pb[bi].c1W), P(pb[bi].c1b), buf1, S_, FNUM_, FNUM_);
    }
    float* xs = buf1;                   // [8,1200,96]
    // self attention (params sorted: Wk,Wo,Wq,Wv,bk,bo,bq,bv)
    int a = pb[bi].attn;
    gemm(xs, 0, FNUM_, P(a + 2), 0, FNUM_, 0, qh, 0, FNUM_, P(a + 6), B_ * S_, FNUM_, FNUM_, 1.f, 0, 1);
    gemm(xs, 0, FNUM_, P(a + 0), 0, FNUM_, 0, kh, 0, FNUM_, P(a + 4), B_ * S_, FNUM_, FNUM_, 1.f, 0, 1);
    gemm(xs, 0, FNUM_, P(a + 3), 0, FNUM_, 0, vh, 0, FNUM_, P(a + 7), B_ * S_, FNUM_, FNUM_, 1.f, 0, 1);
    gemm(qh, (long long)S_ * FNUM_, FNUM_, kh, (long long)S_ * FNUM_, FNUM_, 1,
         scores, (long long)S_ * S_, S_, nullptr, S_, S_, FNUM_, rs96, 0, B_);
    softmax_rows<<<dim3(S_, B_), 256, 0, stream>>>(scores, mask, S_, S_);
    gemm(scores, (long long)S_ * S_, S_, vh, (long long)S_ * FNUM_, FNUM_, 0,
         ctx, (long long)S_ * FNUM_, FNUM_, nullptr, S_, FNUM_, S_, 1.f, 0, B_);
    gemm(ctx, 0, FNUM_, P(a + 1), 0, FNUM_, 0, xa, 0, FNUM_, P(a + 5), B_ * S_, FNUM_, FNUM_, 1.f, 0, 1);
    // seq = mix(xs, xa) transposed to [B,96,S]
    {
      long long tot = (long long)B_ * FNUM_ * S_;
      mix_transpose<<<(unsigned)((tot + 255) / 256), 256, 0, stream>>>(
          xs, xa, P(pb[bi].mix), seqbs[bi], S_, FNUM_);
    }
    // lin chain: [8,115200] -> 1024 -> 256 -> 96 (relu each)
    int l = pb[bi].lin;
    gemm(seqbs[bi], 0, FNUM_ * S_, P(l + 0), 0, 1024, 0, f1, 0, 1024, P(l + 1),
         B_, 1024, FNUM_ * S_, 1.f, 1, 1);
    gemm(f1, 0, 1024, P(l + 2), 0, 256, 0, f2, 0, 256, P(l + 3), B_, 256, 1024, 1.f, 1, 1);
    gemm(f2, 0, 256, P(l + 4), 0, 96, 0, feats + bi * FNUM_, 0, 192, P(l + 5),
         B_, FNUM_, 256, 1.f, 1, 1);
  }
  // seq_feat = wb0*seq0 + wb1*seq1 ; seqT = transpose
  {
    long long tot = (long long)B_ * FNUM_ * S_;
    wsum2<<<(unsigned)((tot + 255) / 256), 256, 0, stream>>>(seqb0, seqb1, P(377), seqfeat, tot);
    transpose_cs<<<(unsigned)((tot + 255) / 256), 256, 0, stream>>>(seqfeat, seqT, S_, FNUM_);
  }
  // prot_feat = feats @ out_W + out_b -> z[:,192:288]
  gemm(feats, 0, 192, P(373), 0, FNUM_, 0, z + 192, 0, 288, P(374), B_, FNUM_, 192, 1.f, 0, 1);

  // ================= ligand DenseNet-GAT =================
  run_gat(x_in, 22, 22, 32, 311, G1, 456);
  float* G = G1; float* Gn = G2;
  int w0 = 32;
  const int trans_base[3] = {317, 323, 329};
  for (int blk = 0; blk < 3; ++blk) {
    for (int j = 0; j < 8; ++j) {
      int base = 21 + blk * 96 + j * 12;
      run_gat(G, 456, w0 + j * 32, 64, base, gT, 64);            // c1 on concat
      run_gat(gT, 64, 64, 32, base + 6, G + (w0 + j * 32), 456); // c2 appended
    }
    int width = w0 + 256;
    run_gat(G, 456, width, width / 2, trans_base[blk], Gn, 456);
    float* tmp = G; G = Gn; Gn = tmp;
    w0 = width / 2;
  }
  // node_f = g @ cls_W + cls_b   (final width 228)
  gemm(G, 0, 456, P(309), 0, FNUM_, 0, nodef, 0, FNUM_, P(310), N_, FNUM_, 228, 1.f, 0, 1);
  // lig = mean pool per graph (exactly 40 nodes each) -> z[:,96:192]
  mean40<<<B_, FNUM_, 0, stream>>>(nodef, z + 96, 288);

  // ================= cross attention fusion =================
  // cross params: 13 Wk, 14 Wo, 15 Wq, 16 Wv, 17 bk, 18 bo, 19 bq, 20 bv
  gemm(nodef, 0, FNUM_, P(15), 0, FNUM_, 0, qh2, 0, FNUM_, P(19), N_, FNUM_, FNUM_, 1.f, 0, 1);
  gemm(seqT, 0, FNUM_, P(13), 0, FNUM_, 0, kh, 0, FNUM_, P(17), B_ * S_, FNUM_, FNUM_, 1.f, 0, 1);
  gemm(seqT, 0, FNUM_, P(16), 0, FNUM_, 0, vh, 0, FNUM_, P(20), B_ * S_, FNUM_, FNUM_, 1.f, 0, 1);
  gemm(qh2, (long long)MAXN_ * FNUM_, FNUM_, kh, (long long)S_ * FNUM_, FNUM_, 1,
       scores, (long long)MAXN_ * S_, S_, nullptr, MAXN_, S_, FNUM_, rs96, 0, B_);
  softmax_rows<<<dim3(MAXN_, B_), 256, 0, stream>>>(scores, mask, MAXN_, S_);
  gemm(scores, (long long)MAXN_ * S_, S_, vh, (long long)S_ * FNUM_, FNUM_, 0,
       ctx2, (long long)MAXN_ * FNUM_, FNUM_, nullptr, MAXN_, FNUM_, S_, 1.f, 0, B_);
  gemm(ctx2, 0, FNUM_, P(14), 0, FNUM_, 0, ca, 0, FNUM_, P(18), N_, FNUM_, FNUM_, 1.f, 0, 1);
  mean40<<<B_, FNUM_, 0, stream>>>(ca, z, 288);   // pooled -> z[:,0:96]

  // ================= classifier =================
  gemm(z, 0, 288, P(5), 0, 1024, 0, t1, 0, 1024, P(6), B_, 1024, 288, 1.f, 1, 1);
  gemm(t1, 0, 1024, P(7), 0, 1024, 0, t2, 0, 1024, P(8), B_, 1024, 1024, 1.f, 1, 1);
  gemm(t2, 0, 1024, P(9), 0, 256, 0, t3, 0, 256, P(10), B_, 256, 1024, 1.f, 1, 1);
  gemm(t3, 0, 256, P(11), 0, 1, 0, (float*)d_out, 0, 1, P(12), B_, 1, 256, 1.f, 0, 1);
}
